// STAMP_37409165148969
// MI455X (gfx1250) — compile-verified, matrix operand order checked by reading, not run on hardware
//
#include <hip/hip_runtime.h>

#define T_ 200
#define B_ 2048
#define I_ 128
#define H_ 128
#define K_ 4

typedef __attribute__((ext_vector_type(8)))  float   v8f;
typedef __attribute__((ext_vector_type(8)))  __bf16  v8bf;
typedef __attribute__((ext_vector_type(16))) __bf16  v16bf;

union V16U { v16bf v; v8bf h[2]; };

// ---------------------------------------------------------------------------
// Weight conversion f32 -> bf16
// ---------------------------------------------------------------------------
__global__ __launch_bounds__(256) void k_cvt_w(
    const float* __restrict__ w1, const float* __restrict__ w2,
    const float* __restrict__ w3,
    __bf16* __restrict__ w1bf, __bf16* __restrict__ w2bf,
    __bf16* __restrict__ w3bf)
{
    int idx = blockIdx.x * 256 + threadIdx.x;   // grid 192 -> 49152
    if (idx < 16384)            w1bf[idx]          = (__bf16)w1[idx];
    else if (idx < 32768)       w2bf[idx - 16384]  = (__bf16)w2[idx - 16384];
    else if (idx < 49152)       w3bf[idx - 32768]  = (__bf16)w3[idx - 32768];
}

// load one 16x32 bf16 fragment (A row-major MxK, or B stored as NxK) ---------
__device__ __forceinline__ v16bf load_frag(const __bf16* __restrict__ base,
                                           int row, int k0, int hi)
{
    V16U u;
    const __bf16* p = base + (size_t)row * I_ + k0 + hi * 8;
    u.h[0] = *(const v8bf*)(p);
    u.h[1] = *(const v8bf*)(p + 16);
    return u.v;
}

// ---------------------------------------------------------------------------
// x1 = mask(x) @ w1^T + b1   -> stored bf16 [T,B,H]
// block = 256 thr (8 waves), 128 flat rows (same t, 128 consecutive b)
// ---------------------------------------------------------------------------
__global__ __launch_bounds__(256) void k_gemm_x1(
    const float* __restrict__ inp, const int* __restrict__ lens,
    const __bf16* __restrict__ w1bf, const float* __restrict__ w1b,
    __bf16* __restrict__ x1bf)
{
    __shared__ __bf16 As[128 * 136];
    __shared__ int    lenS[128];

    const int rowBase = blockIdx.x * 128;
    const int t     = rowBase >> 11;     // / 2048
    const int bBase = rowBase & 2047;
    const int j     = threadIdx.x;

    if (j < 128) lenS[j] = lens[bBase + j];
    __syncthreads();

    // stage masked A tile, f32 -> bf16
    const float4* inp4 = (const float4*)(inp + (size_t)rowBase * I_);
    #pragma unroll
    for (int it = 0; it < 16; ++it) {
        int idx = j + it * 256;          // 0..4095
        int r   = idx >> 5;
        int c4  = idx & 31;
        float4 v = inp4[(size_t)r * 32 + c4];
        if (t >= lenS[r]) { v.x = 0.f; v.y = 0.f; v.z = 0.f; v.w = 0.f; }
        int o = r * 136 + c4 * 4;
        As[o + 0] = (__bf16)v.x; As[o + 1] = (__bf16)v.y;
        As[o + 2] = (__bf16)v.z; As[o + 3] = (__bf16)v.w;
    }
    __syncthreads();

    const int w    = j >> 5;
    const int lane = j & 31;
    const int hi   = lane >> 4;
    const int mrow = w * 16 + (lane & 15);

    // A fragments for all 4 k-steps (I = 128)
    v16bf afr[4];
    #pragma unroll
    for (int ks = 0; ks < 4; ++ks) {
        V16U u;
        int off = mrow * 136 + ks * 32 + hi * 8;
        u.h[0] = *(const v8bf*)&As[off];
        u.h[1] = *(const v8bf*)&As[off + 16];
        afr[ks] = u.v;
    }

    #pragma unroll
    for (int ct = 0; ct < 8; ++ct) {
        int n = ct * 16 + (lane & 15);
        // batch all B-fragment loads for this column tile into one clause
        v16bf bfr[4];
        #pragma unroll
        for (int ks = 0; ks < 4; ++ks)
            bfr[ks] = load_frag(w1bf, n, ks * 32, hi);
        float bias = w1b[n];
        v8f acc = {bias, bias, bias, bias, bias, bias, bias, bias};
        #pragma unroll
        for (int ks = 0; ks < 4; ++ks)
            acc = __builtin_amdgcn_wmma_f32_16x16x32_bf16(
                false, afr[ks], false, bfr[ks], (short)0, acc, false, false);
        #pragma unroll
        for (int r = 0; r < 8; ++r) {
            int grow = rowBase + w * 16 + r + hi * 8;   // C: M = r + 8*hi
            x1bf[(size_t)grow * H_ + n] = (__bf16)acc[r];
        }
    }
}

// ---------------------------------------------------------------------------
// Per-sequence gathered rows + partial-sum means at labeled timesteps
// one block per b, 128 threads (thread = channel i)
// ---------------------------------------------------------------------------
__global__ __launch_bounds__(128) void k_sums(
    const float* __restrict__ inp, const int* __restrict__ lens,
    __bf16* __restrict__ gbf, __bf16* __restrict__ mbf)
{
    const int b = blockIdx.x, i = threadIdx.x;
    const int len = lens[b];
    int start = len - K_; if (start < 0) start = 0;

    float acc = 0.f;
    float snap[K_] = {0.f, 0.f, 0.f, 0.f};
    float gv[K_]   = {0.f, 0.f, 0.f, 0.f};
    for (int t = 0; t < len; ++t) {
        float v = inp[((size_t)t * B_ + b) * I_ + i];
        acc += v;
        #pragma unroll
        for (int k = 0; k < K_; ++k)
            if (t == start + k) { snap[k] = acc; gv[k] = v; }
    }
    #pragma unroll
    for (int k = 0; k < K_; ++k) {
        int tc = start + k; if (tc > T_ - 1) tc = T_ - 1;
        float m = snap[k] / (float)(tc + 1);
        size_t o = ((size_t)b * K_ + k) * I_ + i;
        gbf[o] = (__bf16)gv[k];
        mbf[o] = (__bf16)m;
    }
}

// ---------------------------------------------------------------------------
// x2_lab = g @ w2^T ; c = x2_lab + m @ w3^T    (rows = B*K = 8192)
// ---------------------------------------------------------------------------
__global__ __launch_bounds__(256) void k_gemm_c(
    const __bf16* __restrict__ gbf, const __bf16* __restrict__ mbf,
    const __bf16* __restrict__ w2bf, const __bf16* __restrict__ w3bf,
    float* __restrict__ x2lab, float* __restrict__ cbuf)
{
    const int rowBase = blockIdx.x * 128;
    const int j    = threadIdx.x;
    const int w    = j >> 5;
    const int lane = j & 31;
    const int hi   = lane >> 4;
    const int arow = rowBase + w * 16 + (lane & 15);

    // A fragments are column-tile invariant: preload both matrices once
    v16bf gfr[4], mfr[4];
    #pragma unroll
    for (int ks = 0; ks < 4; ++ks) {
        gfr[ks] = load_frag(gbf, arow, ks * 32, hi);
        mfr[ks] = load_frag(mbf, arow, ks * 32, hi);
    }

    #pragma unroll
    for (int ct = 0; ct < 8; ++ct) {
        int n = ct * 16 + (lane & 15);
        v16bf b2[4], b3[4];
        #pragma unroll
        for (int ks = 0; ks < 4; ++ks) {
            b2[ks] = load_frag(w2bf, n, ks * 32, hi);
            b3[ks] = load_frag(w3bf, n, ks * 32, hi);
        }
        v8f acc = {};
        #pragma unroll
        for (int ks = 0; ks < 4; ++ks)
            acc = __builtin_amdgcn_wmma_f32_16x16x32_bf16(
                false, gfr[ks], false, b2[ks], (short)0, acc, false, false);
        #pragma unroll
        for (int r = 0; r < 8; ++r)
            x2lab[(size_t)(rowBase + w * 16 + r + hi * 8) * H_ + n] = acc[r];
        #pragma unroll
        for (int ks = 0; ks < 4; ++ks)
            acc = __builtin_amdgcn_wmma_f32_16x16x32_bf16(
                false, mfr[ks], false, b3[ks], (short)0, acc, false, false);
        #pragma unroll
        for (int r = 0; r < 8; ++r)
            cbuf[(size_t)(rowBase + w * 16 + r + hi * 8) * H_ + n] = acc[r];
    }
}

// ---------------------------------------------------------------------------
// Attention: att[b,k,s] = w0 . sigmoid(x1[s,b]+c[b,k]) ; h = att.x1 + x2_lab
// one block per b; thread j -> (k = j>>6, s-slot = j&63) for att,
//                  (k = j>>6, h0 = (j&63)*2) for the value accumulation
// ---------------------------------------------------------------------------
__global__ __launch_bounds__(256) void k_att(
    const __bf16* __restrict__ x1bf, const int* __restrict__ lens,
    const float* __restrict__ cbuf, const float* __restrict__ x2lab,
    const float* __restrict__ w0, float* __restrict__ out)
{
    __shared__ float x1s[64 * 133];
    __shared__ float attS[K_ * 64];
    __shared__ float cS[K_ * H_];
    __shared__ float w0S[H_];

    const int b = blockIdx.x;
    const int j = threadIdx.x;
    const int len = lens[b];
    int start = len - K_; if (start < 0) start = 0;

    const int  myk = j >> 6;
    const int  h0  = (j & 63) * 2;
    const int  tl  = start + myk;
    const bool lv  = (tl < len);

    cS[j]       = cbuf[(size_t)b * (K_ * H_) + j];
    cS[j + 256] = cbuf[(size_t)b * (K_ * H_) + j + 256];
    if (j < H_) w0S[j] = w0[j];

    float a0 = 0.f, a1 = 0.f;
    for (int s0 = 0; s0 < len; s0 += 64) {
        __syncthreads();
        // stage 64 timesteps of x1 (bf16 -> f32), zero beyond len
        #pragma unroll 8
        for (int it = 0; it < 32; ++it) {
            int idx = j + it * 256;          // 0..8191
            int ls  = idx >> 7;
            int hh  = idx & 127;
            int sg  = s0 + ls;
            float v = (sg < len) ? (float)x1bf[((size_t)sg * B_ + b) * H_ + hh]
                                 : 0.f;
            x1s[ls * 133 + hh] = v;
        }
        __syncthreads();
        // att for my (k, s)
        {
            int ls = j & 63;
            int sg = s0 + ls;
            float a = 0.f;
            if (lv && sg <= tl) {
                const float* cr = &cS[myk * H_];
                #pragma unroll 4
                for (int hh = 0; hh < H_; ++hh) {
                    float v = x1s[ls * 133 + hh] + cr[hh];
                    a += w0S[hh] * (1.0f / (1.0f + __expf(-v)));
                }
            }
            attS[j] = a;      // j == myk*64 + ls
        }
        __syncthreads();
        // accumulate h for my (k, h0..h0+1)
        #pragma unroll 4
        for (int ls = 0; ls < 64; ++ls) {
            float a = attS[myk * 64 + ls];
            a0 += a * x1s[ls * 133 + h0];
            a1 += a * x1s[ls * 133 + h0 + 1];
        }
    }

    size_t o = ((size_t)b * K_ + myk) * H_ + h0;
    if (lv) {
        out[o]     = a0 + x2lab[o];
        out[o + 1] = a1 + x2lab[o + 1];
    } else {
        out[o] = 0.f;
        out[o + 1] = 0.f;
    }
}

// ---------------------------------------------------------------------------
extern "C" void kernel_launch(void* const* d_in, const int* in_sizes, int n_in,
                              void* d_out, int out_size, void* d_ws, size_t ws_size,
                              hipStream_t stream) {
    const float* inp  = (const float*)d_in[0];
    const int*   lens = (const int*)d_in[1];
    // d_in[2] = label_len (compile-time K_=4)
    const float* w0   = (const float*)d_in[3];
    const float* w1   = (const float*)d_in[4];
    const float* w1b  = (const float*)d_in[5];
    const float* w2   = (const float*)d_in[6];
    const float* w3   = (const float*)d_in[7];
    float* out = (float*)d_out;

    char* ws = (char*)d_ws;
    size_t off = 0;
    auto alloc = [&](size_t bytes) -> void* {
        void* p = ws + off;
        off += (bytes + 255) & ~(size_t)255;
        return p;
    };
    __bf16* x1bf  = (__bf16*)alloc((size_t)T_ * B_ * H_ * 2);   // ~100 MB
    __bf16* w1bf  = (__bf16*)alloc((size_t)H_ * I_ * 2);
    __bf16* w2bf  = (__bf16*)alloc((size_t)H_ * I_ * 2);
    __bf16* w3bf  = (__bf16*)alloc((size_t)H_ * I_ * 2);
    __bf16* gbf   = (__bf16*)alloc((size_t)B_ * K_ * I_ * 2);
    __bf16* mbf   = (__bf16*)alloc((size_t)B_ * K_ * I_ * 2);
    float*  x2lab = (float*)alloc((size_t)B_ * K_ * H_ * 4);
    float*  cbuf  = (float*)alloc((size_t)B_ * K_ * H_ * 4);

    k_cvt_w<<<192, 256, 0, stream>>>(w1, w2, w3, w1bf, w2bf, w3bf);
    k_gemm_x1<<<(T_ * B_) / 128, 256, 0, stream>>>(inp, lens, w1bf, w1b, x1bf);
    k_sums<<<B_, 128, 0, stream>>>(inp, lens, gbf, mbf);
    k_gemm_c<<<(B_ * K_) / 128, 256, 0, stream>>>(gbf, mbf, w2bf, w3bf, x2lab, cbuf);
    k_att<<<B_, 256, 0, stream>>>(x1bf, lens, cbuf, x2lab, w0, out);
}
